// RowMLPAggregator_28011776705094
// MI455X (gfx1250) — matrix operand order repaired and checked by reading, not compiled, not run on hardware
//
#include <hip/hip_runtime.h>
#include <hip/hip_bf16.h>

// ---------------------------------------------------------------------------
// RowMLPAggregator on MI455X (gfx1250, wave32, WMMA).
//
// Precision strategy: fp32 values are split into bf16 hi + bf16 lo
// (lo = fp32(v) - fp32(hi)); products computed as hi*hi + lo*hi + hi*lo with
// fp32 WMMA accumulation -> ~2^-18 relative error, fp32-faithful, at bf16
// WMMA rates (8x the f32 16x16x4 WMMA rate).
// ---------------------------------------------------------------------------

#define DIM   128
#define CAT   640           // 5 * DIM
#define NCH1  20            // CAT / 32  K-chunks for GEMM1
#define NCH2  4             // DIM / 32  K-chunks for GEMM2
#define BM    32            // batch rows per workgroup (2 M-tiles of 16)
#define FRAG  512           // elems per 16x32 bf16 fragment (32 lanes * 16)

typedef __attribute__((ext_vector_type(16))) __bf16          v16bf;
typedef __attribute__((ext_vector_type(8)))  float           v8f;
typedef __attribute__((ext_vector_type(4)))  float           f32x4;
typedef __attribute__((ext_vector_type(4)))  unsigned short  u16x4;

static __device__ __forceinline__ unsigned short bf16_hi_bits(float f) {
  unsigned u = __builtin_bit_cast(unsigned, f);
  unsigned r = u + 0x7FFFu + ((u >> 16) & 1u);   // round-to-nearest-even
  return (unsigned short)(r >> 16);
}
static __device__ __forceinline__ float bits_to_f32(unsigned short h) {
  unsigned u = ((unsigned)h) << 16;
  return __builtin_bit_cast(float, u);
}

// A-fragment position for 16-bit 16x32 A tile (ISA 7.12.2):
// lane(0..15) holds M=lane, K = 0..7 (e0..7) and 16..23 (e8..15)
// lane(16..31) holds M=lane-16, K = 8..15 (e0..7) and 24..31 (e8..15)
static __device__ __forceinline__ void a_frag_pos(int mrow, int kk,
                                                  int& lane, int& e) {
  int g = kk >> 3;                        // 0..3
  int high = ((g == 1) | (g == 3));
  lane = mrow + (high ? 16 : 0);
  e = kk - ((g >= 1) ? 8 : 0) - ((g == 3) ? 8 : 0);
}

// ---------------------------------------------------------------------------
// Pre-pack W (fp32 [out, in]) into bf16 hi/lo B-fragments:
// frag[nt][c][lane][e]: N = nt*16 + (lane&15), K = c*32 + (lane>=16?8:0) + (e>=8?8:0) + e
// (16-bit 32x16 B layout, same K order as A: lanes 16..31 hold K+8.)
// ---------------------------------------------------------------------------
__global__ void __launch_bounds__(256) pack_w_bf16x2(
    const float* __restrict__ W, unsigned short* __restrict__ hi,
    unsigned short* __restrict__ lo, int in_dim, int n_chunks, int total) {
  int idx = blockIdx.x * 256 + threadIdx.x;
  if (idx >= total) return;
  int e    = idx & 15;
  int lane = (idx >> 4) & 31;
  int c    = (idx >> 9) % n_chunks;
  int nt   = idx / (FRAG * n_chunks);
  int n = nt * 16 + (lane & 15);
  int k = c * 32 + ((lane >= 16) ? 8 : 0) + ((e >= 8) ? 8 : 0) + e;
  float v = W[n * in_dim + k];
  unsigned short h = bf16_hi_bits(v);
  hi[idx] = h;
  lo[idx] = bf16_hi_bits(v - bits_to_f32(h));
}

// ---------------------------------------------------------------------------
// Fused gather + MLP kernel. One block = 32 batch rows, 8 wave32s.
// ---------------------------------------------------------------------------
__global__ void __launch_bounds__(256) rowmlp_kernel(
    const float* __restrict__ x_rows,
    const int*   __restrict__ seeds,
    const float* __restrict__ b1,
    const float* __restrict__ b2,
    const unsigned short* __restrict__ w1hi,
    const unsigned short* __restrict__ w1lo,
    const unsigned short* __restrict__ w2hi,
    const unsigned short* __restrict__ w2lo,
    float* __restrict__ out,
    int n_rows) {
  __shared__ alignas(32) unsigned short A1hi[2][NCH1][FRAG];  // 40 KB
  __shared__ alignas(32) unsigned short A1lo[2][NCH1][FRAG];  // 40 KB
  __shared__ alignas(32) unsigned short A2hi[2][NCH2][FRAG];  //  8 KB
  __shared__ alignas(32) unsigned short A2lo[2][NCH2][FRAG];  //  8 KB
  __shared__ int srow[BM * 5];

  const int tid  = threadIdx.x;
  const int wave = tid >> 5;     // wave32: 8 waves per block
  const int lane = tid & 31;
  const int row0 = blockIdx.x * BM;

  // Clamped neighbor row indices for the tile.
  if (tid < BM * 5) {
    int m = tid / 5, j = tid % 5;
    int s = seeds[row0 + m] + j - 2;
    s = s < 0 ? 0 : (s >= n_rows ? n_rows - 1 : s);
    srow[tid] = s;
  }
  __syncthreads();

  // Stage gathered x_cat tile (32 x 640 fp32) into WMMA A-fragment order as
  // bf16 hi/lo. 4 consecutive K per thread-iter: one float4 load, two b64
  // LDS stores (a 4-run at k%4==0 never crosses an 8-element lane group).
  for (int i = tid; i < BM * (CAT / 4); i += 256) {
    int m  = i / (CAT / 4);
    int k4 = (i - m * (CAT / 4)) * 4;
    int j   = k4 >> 7;
    int col = k4 & (DIM - 1);
    f32x4 v = *(const f32x4*)(x_rows + (long long)srow[m * 5 + j] * DIM + col);
    int c  = k4 >> 5;
    int kk = k4 & 31;
    int fl, fe;
    a_frag_pos(m & 15, kk, fl, fe);
    int mt = m >> 4;
    u16x4 hh, ll;
#pragma unroll
    for (int q = 0; q < 4; ++q) {
      unsigned short h = bf16_hi_bits(v[q]);
      hh[q] = h;
      ll[q] = bf16_hi_bits(v[q] - bits_to_f32(h));
    }
    *(u16x4*)&A1hi[mt][c][fl * 16 + fe] = hh;
    *(u16x4*)&A1lo[mt][c][fl * 16 + fe] = ll;
  }
  __syncthreads();

  // ---- GEMM1: h = relu(x_cat @ W1^T + b1); wave owns N-tile = wave ----
  v8f acc0 = {};
  v8f acc1 = {};
  const unsigned short* w1h = w1hi + (wave * NCH1) * FRAG + lane * 16;
  const unsigned short* w1l = w1lo + (wave * NCH1) * FRAG + lane * 16;
  for (int c = 0; c < NCH1; ++c) {
    if (c + 1 < NCH1) {
      __builtin_prefetch(w1h + (c + 1) * FRAG, 0, 0);
      __builtin_prefetch(w1l + (c + 1) * FRAG, 0, 0);
    }
    v16bf bh  = *(const v16bf*)(w1h + c * FRAG);
    v16bf bl  = *(const v16bf*)(w1l + c * FRAG);
    v16bf a0h = *(const v16bf*)&A1hi[0][c][lane * 16];
    v16bf a0l = *(const v16bf*)&A1lo[0][c][lane * 16];
    v16bf a1h = *(const v16bf*)&A1hi[1][c][lane * 16];
    v16bf a1l = *(const v16bf*)&A1lo[1][c][lane * 16];
    acc0 = __builtin_amdgcn_wmma_f32_16x16x32_bf16(false, a0h, false, bh, (short)0, acc0, false, false);
    acc0 = __builtin_amdgcn_wmma_f32_16x16x32_bf16(false, a0l, false, bh, (short)0, acc0, false, false);
    acc0 = __builtin_amdgcn_wmma_f32_16x16x32_bf16(false, a0h, false, bl, (short)0, acc0, false, false);
    acc1 = __builtin_amdgcn_wmma_f32_16x16x32_bf16(false, a1h, false, bh, (short)0, acc1, false, false);
    acc1 = __builtin_amdgcn_wmma_f32_16x16x32_bf16(false, a1l, false, bh, (short)0, acc1, false, false);
    acc1 = __builtin_amdgcn_wmma_f32_16x16x32_bf16(false, a1h, false, bl, (short)0, acc1, false, false);
  }

  // Bias + ReLU, then scatter h into A-fragment layout (hi/lo) for GEMM2.
  {
    float bias1 = b1[wave * 16 + (lane & 15)];
    int kcol = wave * 16 + (lane & 15);      // K index of h in GEMM2
    int c2 = kcol >> 5;
    int kk2 = kcol & 31;
    int g2 = kk2 >> 3;
    int high2 = (((g2 == 1) | (g2 == 3)) ? 16 : 0);
    int fe2 = kk2 - ((g2 >= 1) ? 8 : 0) - ((g2 == 3) ? 8 : 0);
    int mbase = (lane >> 4) * 8;             // D-layout: M = r + 8*(lane>=16)
#pragma unroll
    for (int r = 0; r < 8; ++r) {
      int ml = mbase + r;
      float h0 = acc0[r] + bias1; h0 = h0 > 0.f ? h0 : 0.f;
      float h1 = acc1[r] + bias1; h1 = h1 > 0.f ? h1 : 0.f;
      unsigned short q0 = bf16_hi_bits(h0);
      unsigned short q1 = bf16_hi_bits(h1);
      A2hi[0][c2][(ml + high2) * 16 + fe2] = q0;
      A2lo[0][c2][(ml + high2) * 16 + fe2] = bf16_hi_bits(h0 - bits_to_f32(q0));
      A2hi[1][c2][(ml + high2) * 16 + fe2] = q1;
      A2lo[1][c2][(ml + high2) * 16 + fe2] = bf16_hi_bits(h1 - bits_to_f32(q1));
    }
  }
  __syncthreads();

  // ---- GEMM2: out = h @ W2^T + b2 ----
  v8f o0 = {};
  v8f o1 = {};
  const unsigned short* w2h = w2hi + (wave * NCH2) * FRAG + lane * 16;
  const unsigned short* w2l = w2lo + (wave * NCH2) * FRAG + lane * 16;
#pragma unroll
  for (int c = 0; c < NCH2; ++c) {
    v16bf bh  = *(const v16bf*)(w2h + c * FRAG);
    v16bf bl  = *(const v16bf*)(w2l + c * FRAG);
    v16bf a0h = *(const v16bf*)&A2hi[0][c][lane * 16];
    v16bf a0l = *(const v16bf*)&A2lo[0][c][lane * 16];
    v16bf a1h = *(const v16bf*)&A2hi[1][c][lane * 16];
    v16bf a1l = *(const v16bf*)&A2lo[1][c][lane * 16];
    o0 = __builtin_amdgcn_wmma_f32_16x16x32_bf16(false, a0h, false, bh, (short)0, o0, false, false);
    o0 = __builtin_amdgcn_wmma_f32_16x16x32_bf16(false, a0l, false, bh, (short)0, o0, false, false);
    o0 = __builtin_amdgcn_wmma_f32_16x16x32_bf16(false, a0h, false, bl, (short)0, o0, false, false);
    o1 = __builtin_amdgcn_wmma_f32_16x16x32_bf16(false, a1h, false, bh, (short)0, o1, false, false);
    o1 = __builtin_amdgcn_wmma_f32_16x16x32_bf16(false, a1l, false, bh, (short)0, o1, false, false);
    o1 = __builtin_amdgcn_wmma_f32_16x16x32_bf16(false, a1h, false, bl, (short)0, o1, false, false);
  }

  {
    float bias2 = b2[wave * 16 + (lane & 15)];
    int colo = wave * 16 + (lane & 15);
    int mbase = (lane >> 4) * 8;
#pragma unroll
    for (int r = 0; r < 8; ++r) {
      out[(long long)(row0 + mbase + r) * DIM + colo]      = o0[r] + bias2;
      out[(long long)(row0 + 16 + mbase + r) * DIM + colo] = o1[r] + bias2;
    }
  }
}

// ---------------------------------------------------------------------------
extern "C" void kernel_launch(void* const* d_in, const int* in_sizes, int n_in,
                              void* d_out, int out_size, void* d_ws, size_t ws_size,
                              hipStream_t stream) {
  const float* x_rows = (const float*)d_in[0];
  const int*   seeds  = (const int*)d_in[1];
  const float* W1     = (const float*)d_in[2];
  const float* b1     = (const float*)d_in[3];
  const float* W2     = (const float*)d_in[4];
  const float* b2     = (const float*)d_in[5];
  float* out = (float*)d_out;

  const int n_rows = in_sizes[0] / DIM;   // 1,000,000
  const int batch  = in_sizes[1];         // 262,144

  const int w1_elems = 8 * NCH1 * FRAG;   // 81920
  const int w2_elems = 8 * NCH2 * FRAG;   // 16384
  unsigned short* w1hi = (unsigned short*)d_ws;
  unsigned short* w1lo = w1hi + w1_elems;
  unsigned short* w2hi = w1lo + w1_elems;
  unsigned short* w2lo = w2hi + w2_elems;

  pack_w_bf16x2<<<(w1_elems + 255) / 256, 256, 0, stream>>>(
      W1, w1hi, w1lo, CAT, NCH1, w1_elems);
  pack_w_bf16x2<<<(w2_elems + 255) / 256, 256, 0, stream>>>(
      W2, w2hi, w2lo, DIM, NCH2, w2_elems);

  rowmlp_kernel<<<batch / BM, 256, 0, stream>>>(
      x_rows, seeds, b1, b2, w1hi, w1lo, w2hi, w2lo, out, n_rows);
}
